// RecurrentClassifierOperator_25391846654171
// MI455X (gfx1250) — matrix operand order repaired and checked by reading, not compiled, not run on hardware
//
#include <hip/hip_runtime.h>

// ---------------- problem constants ----------------
static constexpr int B    = 1024;
static constexpr int S    = 512;
static constexpr int D    = 128;
static constexpr int H    = 128;
static constexpr int G    = 4 * H;   // 512 gate outputs
static constexpr int OPS  = 16;
static constexpr int ROWS = 16;      // batch rows per workgroup (WMMA M)
static constexpr int SXP  = 264;     // padded LDS row stride for [x|h] activations (bf16 elems)

// ---------------- CDNA5 WMMA types ----------------
typedef __attribute__((ext_vector_type(16))) __bf16 v16bf;
typedef __attribute__((ext_vector_type(8)))  float  v8f;
typedef unsigned int v4u __attribute__((ext_vector_type(4)));

union ABFrag { v16bf v; v4u q[2]; unsigned short us[16]; };

__device__ __forceinline__ unsigned short f2bf(float f) {
  unsigned u = __builtin_bit_cast(unsigned, f);
  u += 0x7FFFu + ((u >> 16) & 1u);       // round-to-nearest-even
  return (unsigned short)(u >> 16);
}

__device__ __forceinline__ unsigned pack2bf(float a, float b) {
  return (unsigned)f2bf(a) | ((unsigned)f2bf(b) << 16);
}

// native gfx1250 v_tanh_f32 when the toolchain exposes it
#if __has_builtin(__builtin_amdgcn_tanhf)
__device__ __forceinline__ float fast_tanh(float x) { return __builtin_amdgcn_tanhf(x); }
#else
__device__ __forceinline__ float fast_tanh(float x) { return tanhf(x); }
#endif
__device__ __forceinline__ float fast_sigmoid(float x) {
  return __builtin_fmaf(0.5f, fast_tanh(0.5f * x), 0.5f);
}

// ===================================================================
// Fused LSTM: one workgroup = 16 batch rows, 8 waves.
// Weights live as bf16 WMMA B-fragments in registers (256 VGPRs/lane,
// loop-invariant).  Per step: load A ([x|h], 16x256 bf16) from LDS,
// 32 back-to-back v_wmma_f32_16x16x32_bf16, LSTM update in the
// accumulator layout, h written back to LDS for the next step.
// ===================================================================
__global__ void __launch_bounds__(256, 1)
lstm_fused_kernel(const int*   __restrict__ tokens,     // [B,S]
                  const int*   __restrict__ lengths,    // [B]
                  const float* __restrict__ embedding,  // [V2,D]
                  const float* __restrict__ W_ih,       // [G,D]
                  const float* __restrict__ W_hh,       // [G,H]
                  const float* __restrict__ b_ih,       // [G]
                  const float* __restrict__ b_hh,       // [G]
                  const float* __restrict__ h0,         // [H]
                  const float* __restrict__ c0,         // [H]
                  float*       __restrict__ hlast)      // [B,H] workspace
{
  __shared__ unsigned short sX[ROWS * SXP];   // [x(0..127) | h(128..255)] bf16, padded rows
  __shared__ float          sBias[G];
  __shared__ int            sLen[ROWS];

  const int tid  = threadIdx.x;
  const int lane = tid & 31;
  const int wv   = tid >> 5;               // wave 0..7
  const int b0   = blockIdx.x * ROWS;

  const int col16 = lane & 15;
  const int hcol  = wv * 16 + col16;       // hidden column owned by this (wave,lane)
  const int mbase = (lane < 16) ? 0 : 8;

  // ---- preamble: biases, lengths, initial h into LDS ----
  for (int idx = tid; idx < G; idx += 256) sBias[idx] = b_ih[idx] + b_hh[idx];
  if (tid < ROWS) { int L = lengths[b0 + tid]; sLen[tid] = (L < 1 ? 1 : L) - 1; }
  for (int idx = tid; idx < ROWS * H; idx += 256) {
    int m = idx >> 7, k = idx & 127;
    sX[m * SXP + D + k] = f2bf(h0[k]);     // initial hidden state
  }

  // ---- preamble: register-stationary B fragments (bf16) from global f32 ----
  // B-matrix (32x16) layout: lanes 0-15 hold K 0..15, lanes 16-31 hold K 16..31,
  // VGPR j = K pair (2j, 2j+1).  Lane 'col16' carries weight row (gate output).
  const int b_koff = (lane < 16) ? 0 : 16;
  ABFrag Wb[4][8];                          // [gate i/f/g/o][k-slice] -> 256 VGPRs
  #pragma unroll
  for (int g = 0; g < 4; ++g) {
    const int row = (g * 8 + wv) * 16 + col16;     // gate-output row in [0,512)
    #pragma unroll
    for (int kk = 0; kk < 8; ++kk) {
      const int kb0 = kk * 32 + b_koff;            // 16-elem chunk start, never straddles D
      const float* src = (kk < 4) ? (W_ih + (size_t)row * D + kb0)
                                  : (W_hh + (size_t)row * H + (kb0 - D));
      #pragma unroll
      for (int q = 0; q < 4; ++q) {
        float4 f = *(const float4*)(src + q * 4);
        unsigned lo = pack2bf(f.x, f.y);
        unsigned hi = pack2bf(f.z, f.w);
        Wb[g][kk].q[q >> 1][2 * (q & 1) + 0] = lo;
        Wb[g][kk].q[q >> 1][2 * (q & 1) + 1] = hi;
      }
    }
  }
  __syncthreads();

  const float bias_i = sBias[          hcol];
  const float bias_f = sBias[    H  +  hcol];
  const float bias_g = sBias[2 * H  +  hcol];
  const float bias_o = sBias[3 * H  +  hcol];

  v8f c;                                   // persistent cell state (f32)
  {
    float cv = c0[hcol];
    #pragma unroll
    for (int r = 0; r < 8; ++r) c[r] = cv;
  }

  // A-fragment addressing (16x32 bf16): lanes 0-15 -> K 0-15,
  // lanes 16-31 -> K 16-31, per 32-wide k-slice.
  const int a_row  = (lane & 15) * SXP;
  const int a_koff = (lane < 16) ? 0 : 16;

  // x-gather assignment: thread -> (row m, 8 consecutive k)
  const int gm = tid >> 4;                 // 0..15
  const int gk = (tid & 15) * 8;           // 0..120

  for (int t = 0; t < S; ++t) {
    // ---- phase 1: gather x_t with ReLU, pack to bf16, one b128 LDS store ----
    {
      int tok = tokens[(b0 + gm) * S + t];
      const float* er = embedding + (size_t)tok * D + gk;
      float4 f0 = *(const float4*)(er);
      float4 f1 = *(const float4*)(er + 4);
      v4u pk;
      pk[0] = pack2bf(fmaxf(f0.x, 0.f), fmaxf(f0.y, 0.f));
      pk[1] = pack2bf(fmaxf(f0.z, 0.f), fmaxf(f0.w, 0.f));
      pk[2] = pack2bf(fmaxf(f1.x, 0.f), fmaxf(f1.y, 0.f));
      pk[3] = pack2bf(fmaxf(f1.z, 0.f), fmaxf(f1.w, 0.f));
      *(v4u*)(&sX[gm * SXP + gk]) = pk;
      if ((t + 1) < S && (tid & 15) == 0) {
        int tok2 = tokens[(b0 + gm) * S + t + 1];
        __builtin_prefetch(&embedding[(size_t)tok2 * D], 0, 0);  // global_prefetch_b8
      }
    }
    __syncthreads();   // x ready; previous step's h writes visible

    // ---- phase 2: preload full-K A fragments ([x|h]) ----
    ABFrag a[8];
    #pragma unroll
    for (int kk = 0; kk < 8; ++kk) {
      int kb = kk * 32 + a_koff;
      a[kk].q[0] = *(const v4u*)(&sX[a_row + kb]);
      a[kk].q[1] = *(const v4u*)(&sX[a_row + kb + 8]);
    }
    __syncthreads();   // all waves done reading old h before it is overwritten

    // ---- phase 3: gates = [x|h] @ Wc^T, all operands in registers ----
    v8f acc_i = {}, acc_f = {}, acc_g = {}, acc_o = {};
    #pragma unroll
    for (int kk = 0; kk < 8; ++kk) {
      acc_i = __builtin_amdgcn_wmma_f32_16x16x32_bf16(false, a[kk].v, false, Wb[0][kk].v, (short)0, acc_i, false, false);
      acc_f = __builtin_amdgcn_wmma_f32_16x16x32_bf16(false, a[kk].v, false, Wb[1][kk].v, (short)0, acc_f, false, false);
      acc_g = __builtin_amdgcn_wmma_f32_16x16x32_bf16(false, a[kk].v, false, Wb[2][kk].v, (short)0, acc_g, false, false);
      acc_o = __builtin_amdgcn_wmma_f32_16x16x32_bf16(false, a[kk].v, false, Wb[3][kk].v, (short)0, acc_o, false, false);
    }

    // ---- LSTM elementwise update in accumulator layout ----
    #pragma unroll
    for (int r = 0; r < 8; ++r) {
      float ig = fast_sigmoid(acc_i[r] + bias_i);
      float fg = fast_sigmoid(acc_f[r] + bias_f);
      float gg = fast_tanh(acc_g[r] + bias_g);
      float og = fast_sigmoid(acc_o[r] + bias_o);
      float cv = __builtin_fmaf(fg, c[r], ig * gg);
      float hv = og * fast_tanh(cv);
      c[r] = cv;
      int m = mbase + r;
      sX[m * SXP + D + hcol] = f2bf(hv);                  // feed h back for next step
      if (t == sLen[m]) hlast[(b0 + m) * H + hcol] = hv;  // ragged last-step capture
    }
    // next iteration's first barrier orders these h writes vs. reads
  }
}

// ===================================================================
// Head: out[b,j] = relu(h_last[b]) . lin_W[j,:H] + onehot[b] . lin_W[j,H:] + lin_b[j]
// ===================================================================
__global__ void head_kernel(const float* __restrict__ hlast,
                            const float* __restrict__ onehot,
                            const float* __restrict__ lin_W,   // [2, H+OPS]
                            const float* __restrict__ lin_b,   // [2]
                            float*       __restrict__ out)     // [B, 2]
{
  int gid = blockIdx.x * blockDim.x + threadIdx.x;
  if (gid >= B * 2) return;
  int b = gid >> 1, j = gid & 1;
  const float* w = lin_W + j * (H + OPS);
  float acc = lin_b[j];
  #pragma unroll 4
  for (int k = 0; k < H; ++k) {
    float hv = hlast[b * H + k];
    acc += (hv > 0.0f ? hv : 0.0f) * w[k];
  }
  #pragma unroll
  for (int k = 0; k < OPS; ++k) acc += onehot[b * OPS + k] * w[H + k];
  out[gid] = acc;
}

// ===================================================================
extern "C" void kernel_launch(void* const* d_in, const int* in_sizes, int n_in,
                              void* d_out, int out_size, void* d_ws, size_t ws_size,
                              hipStream_t stream) {
  (void)in_sizes; (void)n_in; (void)out_size; (void)ws_size;
  const int*   tokens    = (const int*)d_in[0];
  const int*   lengths   = (const int*)d_in[1];
  const float* onehot    = (const float*)d_in[2];
  const float* embedding = (const float*)d_in[3];
  const float* W_ih      = (const float*)d_in[4];
  const float* W_hh      = (const float*)d_in[5];
  const float* b_ih      = (const float*)d_in[6];
  const float* b_hh      = (const float*)d_in[7];
  const float* h0        = (const float*)d_in[8];
  const float* c0        = (const float*)d_in[9];
  const float* lin_W     = (const float*)d_in[10];
  const float* lin_b     = (const float*)d_in[11];
  float* hlast = (float*)d_ws;                       // B*H floats = 512 KB
  float* out   = (float*)d_out;

  lstm_fused_kernel<<<B / ROWS, 256, 0, stream>>>(
      tokens, lengths, embedding, W_ih, W_hh, b_ih, b_hh, h0, c0, hlast);

  head_kernel<<<(B * 2 + 255) / 256, 256, 0, stream>>>(hlast, onehot, lin_W, lin_b, out);
}
